// DSConv_2430951489992
// MI455X (gfx1250) — matrix operand order repaired and checked by reading, not compiled, not run on hardware
//
#include <hip/hip_runtime.h>
#include <hip/hip_bf16.h>
#include <math.h>

#define BATCH 2
#define CIN   64
#define COUT  64
#define KK    9
#define WW    192
#define HH    192
#define WH    (WW*HH)        // 36864
#define NPIX  (BATCH*WH)     // 73728
#define KTOT  (KK*CIN)       // 576
#define EPSV  1e-5f

#define WSTRIDE 584          // padded weight row (halves): 576 + 8
#define XROW    72           // padded X row (halves): 64 + 8
#define PIXB    256          // pixels per block in the GEMM kernel

typedef _Float16 v8h  __attribute__((ext_vector_type(8)));
typedef _Float16 v16h __attribute__((ext_vector_type(16)));
typedef float    v8f  __attribute__((ext_vector_type(8)));

// ---- workspace layout (in floats) ----
#define FEATT_OFF 0
#define FEATT_SZ  (WH*CIN)                    // 2,359,296 floats
#define TBLW_OFF  (FEATT_OFF + FEATT_SZ)
#define TBL_SZ    (NPIX*KK*4)                 // 2,654,208 floats
#define TBLI_OFF  (TBLW_OFF + TBL_SZ)
#define Z_OFF     (TBLI_OFF + TBL_SZ)
#define Z_SZ      (BATCH*COUT*WH)             // 4,718,592 floats
#define STAT_OFF  (Z_OFF + Z_SZ)              // 64 floats
#define WF16_OFF  (STAT_OFF + 64)             // 64*584 halves = 74,752 bytes

__device__ __forceinline__ int iclamp(int v, int lo, int hi) {
    return v < lo ? lo : (v > hi ? hi : v);
}

// ---------------- Kernel 1: transpose f[0] to pixel-major featT ----------------
__global__ void __launch_bounds__(256) featT_kernel(const float* __restrict__ f,
                                                    float* __restrict__ featT) {
    int t = blockIdx.x * 256 + threadIdx.x;
    if (t >= WH * CIN) return;
    int ci = t / WH, pix = t % WH;
    featT[pix * CIN + ci] = f[ci * WH + pix];   // reference samples f[0] for ALL batches
}

// ---------------- Kernel 2: pre-swizzle conv weights to padded f16 matrix ----------------
// wf16[co][pos], pos = k*64 + ci (pad halves zeroed)
__global__ void __launch_bounds__(256) wprep_kernel(const float* __restrict__ convw,
                                                    _Float16* __restrict__ wf16) {
    int i = blockIdx.x * 256 + threadIdx.x;
    if (i >= COUT * WSTRIDE) return;
    int co = i / WSTRIDE, r = i % WSTRIDE;
    _Float16 v = (_Float16)0.0f;
    if (r < KTOT) {
        int k = r >> 6, ci = r & 63;
        v = (_Float16)convw[(co * CIN + ci) * KK + k];
    }
    wf16[i] = v;
}

// ---------------- Kernel 3: offset conv + tanh + cumsum + sample table ----------------
__global__ void __launch_bounds__(256) offset_kernel(const float* __restrict__ f,
                                                     const float* __restrict__ offw,
                                                     const float* __restrict__ offb,
                                                     const float* __restrict__ bng,
                                                     const float* __restrict__ bnb,
                                                     float4* __restrict__ tblw,
                                                     int4*   __restrict__ tbli) {
    __shared__ float wsm[KK * CIN * 9];  // first 9 output channels only (rest unused)
    __shared__ float sc[KK], sh[KK];
    int tid = threadIdx.x;
    for (int i = tid; i < KK * CIN * 9; i += 256) wsm[i] = offw[i];
    if (tid < KK) {
        float invs = rsqrtf(1.0f + EPSV);
        sc[tid] = invs * bng[tid];
        sh[tid] = offb[tid] * invs * bng[tid] + bnb[tid];
    }
    __syncthreads();

    int t = blockIdx.x * 256 + tid;
    if (t >= NPIX) return;
    int b = t / WH, rem = t % WH, w = rem / HH, h = rem % HH;

    float acc[KK];
#pragma unroll
    for (int oc = 0; oc < KK; ++oc) acc[oc] = 0.0f;

    const float* fb = f + (size_t)b * CIN * WH;
    for (int ci = 0; ci < CIN; ++ci) {
        float tap[9];
#pragma unroll
        for (int dy = 0; dy < 3; ++dy)
#pragma unroll
            for (int dx = 0; dx < 3; ++dx) {
                int y = w + dy - 1, x = h + dx - 1;
                bool ok = (y >= 0) & (y < WW) & (x >= 0) & (x < HH);
                tap[dy * 3 + dx] = ok ? fb[ci * WH + y * HH + x] : 0.0f;
            }
#pragma unroll
        for (int oc = 0; oc < KK; ++oc) {
            const float* wp = &wsm[(oc * CIN + ci) * 9];
#pragma unroll
            for (int u = 0; u < 9; ++u) acc[oc] = fmaf(tap[u], wp[u], acc[oc]);
        }
    }

    float yoff[KK];
#pragma unroll
    for (int oc = 0; oc < KK; ++oc) yoff[oc] = tanhf(acc[oc] * sc[oc] + sh[oc]);

    // exact reference cumsum quirk: center zeroed, j runs 1..3, ends stay raw
    float cum[KK];
    cum[0] = yoff[0]; cum[8] = yoff[8]; cum[4] = 0.0f;
    cum[5] = yoff[5];           cum[6] = cum[5] + yoff[6]; cum[7] = cum[6] + yoff[7];
    cum[3] = yoff[3];           cum[2] = cum[3] + yoff[2]; cum[1] = cum[2] + yoff[1];

#pragma unroll
    for (int k = 0; k < KK; ++k) {
        float yf = (float)w + cum[k];
        float xf = (float)(h + k - 4);
        float y0f = floorf(yf), x0f = floorf(xf);
        int y0i = iclamp((int)y0f,     0, WW - 1);
        int y1i = iclamp((int)y0f + 1, 0, WW - 1);
        int x0i = iclamp((int)x0f,     0, HH - 1);
        int x1i = iclamp((int)x0f + 1, 0, HH - 1);
        float y0w = fminf(fmaxf(y0f,        0.f), (float)WW);
        float y1w = fminf(fmaxf(y0f + 1.f,  0.f), (float)WW);
        float x0w = fminf(fmaxf(x0f,        0.f), (float)HH);
        float x1w = fminf(fmaxf(x0f + 1.f,  0.f), (float)HH);
        float wy1 = y1w - yf, wy0 = yf - y0w;
        float wx1 = x1w - xf, wx0 = xf - x0w;
        tblw[t * KK + k] = make_float4(wy1 * wx1, wy1 * wx0, wy0 * wx1, wy0 * wx0);
        tbli[t * KK + k] = make_int4(y0i * HH + x0i, y0i * HH + x1i,
                                     y1i * HH + x0i, y1i * HH + x1i);
    }
}

// ---------------- Kernel 4: fused gather + WMMA GEMM (z = W[64x576] * X[576xN]) ----------------
// 256 threads = 8 waves; 256 pixels per block. Per k: one thread builds one
// pixel's full 64-ci sampled row into LDS; each wave then runs 2 K-chunks x
// (4 M-tiles x 2 N-tiles) = 16 v_wmma_f32_16x16x32_f16 with A reuse.
__global__ void __launch_bounds__(256) conv_wmma_kernel(const _Float16* __restrict__ wf16,
                                                        const float* __restrict__ convb,
                                                        const float* __restrict__ featT,
                                                        const float4* __restrict__ tblw,
                                                        const int4*  __restrict__ tbli,
                                                        float* __restrict__ z) {
    extern __shared__ _Float16 smem[];
    _Float16* wlds = smem;                    // [64][WSTRIDE]  74,752 B (LDS offset 0)
    _Float16* xlds = smem + COUT * WSTRIDE;   // [256][XROW]    36,864 B

    int tid = threadIdx.x;

    // ---- async bulk copy of pre-swizzled f16 weights: global -> LDS ----
    // 74,752 B = 4672 x b128; dynamic LDS base is byte 0 in this kernel.
    {
        const char* g = (const char*)wf16;
#pragma unroll
        for (int it = 0; it < 18; ++it) {
            unsigned off = (unsigned)(tid + it * 256) * 16u;
            unsigned long long ga = (unsigned long long)(g + off);
            asm volatile("global_load_async_to_lds_b128 %0, %1, off"
                         :: "v"(off), "v"(ga) : "memory");
        }
        if (tid < 64) {
            unsigned off = (unsigned)(tid + 18 * 256) * 16u;
            unsigned long long ga = (unsigned long long)(g + off);
            asm volatile("global_load_async_to_lds_b128 %0, %1, off"
                         :: "v"(off), "v"(ga) : "memory");
        }
        asm volatile("s_wait_asynccnt 0x0" ::: "memory");
    }
    __syncthreads();

    v8f acc[4][2];
#pragma unroll
    for (int m = 0; m < 4; ++m) {
        acc[m][0] = (v8f)0.0f;
        acc[m][1] = (v8f)0.0f;
    }

    int lane = tid & 31, wave = tid >> 5;
    int m15 = lane & 15, hi = lane >> 4;
    int pbuild = blockIdx.x * PIXB + tid;     // pixel this thread gathers

    for (int k = 0; k < KK; ++k) {
        // ---- build X tile: bilinear gather of full 64-ci row per pixel ----
        {
            float4 wv = tblw[pbuild * KK + k];
            int4   iv = tbli[pbuild * KK + k];
            const float4* pa0 = (const float4*)(featT + iv.x * CIN);
            const float4* pc0 = (const float4*)(featT + iv.y * CIN);
            const float4* pa1 = (const float4*)(featT + iv.z * CIN);
            const float4* pc1 = (const float4*)(featT + iv.w * CIN);
            _Float16* xr = xlds + tid * XROW;
#pragma unroll
            for (int q = 0; q < 16; q += 2) {
                v8h hh;
#pragma unroll
                for (int j = 0; j < 2; ++j) {
                    float4 a0 = pa0[q + j], c0 = pc0[q + j];
                    float4 a1 = pa1[q + j], c1 = pc1[q + j];
                    hh[4*j+0] = (_Float16)(wv.x*a0.x + wv.y*c0.x + wv.z*a1.x + wv.w*c1.x);
                    hh[4*j+1] = (_Float16)(wv.x*a0.y + wv.y*c0.y + wv.z*a1.y + wv.w*c1.y);
                    hh[4*j+2] = (_Float16)(wv.x*a0.z + wv.y*c0.z + wv.z*a1.z + wv.w*c1.z);
                    hh[4*j+3] = (_Float16)(wv.x*a0.w + wv.y*c0.w + wv.z*a1.w + wv.w*c1.w);
                }
                *(v8h*)(xr + q * 4) = hh;
            }
        }
        __syncthreads();

#pragma unroll
        for (int h2 = 0; h2 < 2; ++h2) {      // two 32-wide K chunks per k
            int c = 2 * k + h2;
            // B fragments (32x16 f16): lanes 0-15 K=0..15, lanes 16-31 K=16..31
            v16h bf[2];
#pragma unroll
            for (int nt = 0; nt < 2; ++nt) {
                const _Float16* xb = xlds + (wave * 32 + nt * 16 + m15) * XROW
                                     + h2 * 32 + hi * 16;
                v8h b0 = *(const v8h*)(xb);
                v8h b1 = *(const v8h*)(xb + 8);
                bf[nt] = __builtin_shufflevector(b0, b1,
                          0,1,2,3,4,5,6,7,8,9,10,11,12,13,14,15);
            }
#pragma unroll
            for (int m = 0; m < 4; ++m) {
                // A fragment (16x32 f16): lanes 0-15 K=0..15, lanes 16-31 K=16..31
                const _Float16* wa = wlds + (m * 16 + m15) * WSTRIDE + c * 32;
                v8h a0 = *(const v8h*)(wa + hi * 16);
                v8h a1 = *(const v8h*)(wa + hi * 16 + 8);
                v16h af = __builtin_shufflevector(a0, a1,
                           0,1,2,3,4,5,6,7,8,9,10,11,12,13,14,15);
                acc[m][0] = __builtin_amdgcn_wmma_f32_16x16x32_f16(
                    false, af, false, bf[0], (short)0, acc[m][0], false, false);
                acc[m][1] = __builtin_amdgcn_wmma_f32_16x16x32_f16(
                    false, af, false, bf[1], (short)0, acc[m][1], false, false);
            }
        }
        __syncthreads();
    }

    // ---- epilogue: add bias, store z (lane = column/pixel; vgpr r = row/co) ----
#pragma unroll
    for (int nt = 0; nt < 2; ++nt) {
        int p = blockIdx.x * PIXB + wave * 32 + nt * 16 + m15;
        int b = p / WH, rem = p % WH;
#pragma unroll
        for (int m = 0; m < 4; ++m)
#pragma unroll
            for (int r = 0; r < 8; ++r) {
                int co = m * 16 + hi * 8 + r;
                z[(b * COUT + co) * WH + rem] = acc[m][nt][r] + convb[co];
            }
    }
}

// ---------------- Kernel 5: GroupNorm stats (deterministic tree reduction) ----------------
__global__ void __launch_bounds__(256) gn_stats_kernel(const float* __restrict__ z,
                                                       float* __restrict__ stats) {
    __shared__ float ssum[256], ssq[256];
    int b = blockIdx.x >> 4, g = blockIdx.x & 15;
    const float* base = z + (size_t)(b * COUT + g * 4) * WH;
    float s = 0.f, q = 0.f;
    for (int i = threadIdx.x; i < 4 * WH; i += 256) {
        float v = base[i];
        s += v; q += v * v;
    }
    ssum[threadIdx.x] = s; ssq[threadIdx.x] = q;
    __syncthreads();
    for (int off = 128; off > 0; off >>= 1) {
        if (threadIdx.x < off) {
            ssum[threadIdx.x] += ssum[threadIdx.x + off];
            ssq[threadIdx.x]  += ssq[threadIdx.x + off];
        }
        __syncthreads();
    }
    if (threadIdx.x == 0) {
        float n = 4.0f * WH;
        float mu = ssum[0] / n;
        float var = ssq[0] / n - mu * mu;
        stats[blockIdx.x * 2 + 0] = mu;
        stats[blockIdx.x * 2 + 1] = rsqrtf(var + EPSV);
    }
}

// ---------------- Kernel 6: GN apply + ReLU ----------------
__global__ void __launch_bounds__(256) gn_apply_kernel(const float* __restrict__ z,
                                                       const float* __restrict__ stats,
                                                       const float* __restrict__ gng,
                                                       const float* __restrict__ gnb,
                                                       float* __restrict__ out) {
    int i = blockIdx.x * 256 + threadIdx.x;
    if (i >= BATCH * COUT * WH) return;
    int b = i / (COUT * WH);
    int co = (i / WH) % COUT;
    int sg = (b * 16 + (co >> 2)) * 2;
    float mu = stats[sg], rstd = stats[sg + 1];
    float v = (z[i] - mu) * rstd * gng[co] + gnb[co];
    out[i] = fmaxf(v, 0.f);
}

extern "C" void kernel_launch(void* const* d_in, const int* in_sizes, int n_in,
                              void* d_out, int out_size, void* d_ws, size_t ws_size,
                              hipStream_t stream) {
    const float* f     = (const float*)d_in[0];
    const float* offw  = (const float*)d_in[1];
    const float* offb  = (const float*)d_in[2];
    const float* bng   = (const float*)d_in[3];
    const float* bnb   = (const float*)d_in[4];
    const float* convw = (const float*)d_in[5];
    const float* convb = (const float*)d_in[6];
    const float* gng   = (const float*)d_in[7];
    const float* gnb   = (const float*)d_in[8];

    float*     ws    = (float*)d_ws;
    float*     featT = ws + FEATT_OFF;
    float4*    tblw  = (float4*)(ws + TBLW_OFF);
    int4*      tbli  = (int4*)(ws + TBLI_OFF);
    float*     z     = ws + Z_OFF;
    float*     stats = ws + STAT_OFF;
    _Float16*  wf16  = (_Float16*)(ws + WF16_OFF);
    float*     out   = (float*)d_out;

    featT_kernel<<<(WH * CIN + 255) / 256, 256, 0, stream>>>(f, featT);
    wprep_kernel<<<(COUT * WSTRIDE + 255) / 256, 256, 0, stream>>>(convw, wf16);
    offset_kernel<<<NPIX / 256, 256, 0, stream>>>(f, offw, offb, bng, bnb, tblw, tbli);

    size_t smem = (size_t)(COUT * WSTRIDE + PIXB * XROW) * sizeof(_Float16); // ~109 KB
    conv_wmma_kernel<<<NPIX / PIXB, 256, smem, stream>>>(wf16, convb, featT, tblw, tbli, z);

    gn_stats_kernel<<<32, 256, 0, stream>>>(z, stats);
    gn_apply_kernel<<<(BATCH * COUT * WH + 255) / 256, 256, 0, stream>>>(z, stats, gng, gnb, out);
}
